// Original_57956288692241
// MI455X (gfx1250) — matrix operand order repaired and checked by reading, not compiled, not run on hardware
//
#include <hip/hip_runtime.h>
#include <hip/hip_bf16.h>

// ---------------------------------------------------------------------------
// GPT forward (B=4, T=768, D=512, L=6, H=8, hd=64, F=2048, V=50304) on
// MI455X / gfx1250.  All GEMMs run through v_wmma_f32_16x16x32_f16 (fp16 in,
// fp32 accumulate).  Attention scores (75MB f32) + probs (38MB f16) are
// materialized in workspace -- they fit in the 192MB L2, so the round trip
// stays on-chip.  Dominant HBM traffic is the 618MB logits store (~27us @
// 23.3TB/s); everything else is small.  Epilogue variants are compile-time
// template flags so the hot stores are branch-free.
// Workspace requirement: ~232 MB.
// ---------------------------------------------------------------------------

typedef _Float16 v16h __attribute__((ext_vector_type(16)));
typedef _Float16 v8h  __attribute__((ext_vector_type(8)));
typedef float    v8f  __attribute__((ext_vector_type(8)));

// ---------------- WMMA GEMM: C = act(A @ W^T [+ bias] [+ resid]) -----------
// A: [M,K] f16 row-major (row stride K).  W: [N,K] f16 row-major (row stride
// K) -- torch-style [out,in] weight, computing x @ W^T directly.
// One wave per 64x64 output tile, 4x4 grid of 16x16x32 WMMAs per K-step.
// Requires M%64==0, N%64==0, K%32==0 (true for every GEMM in this model).
// Batched over blockIdx.z with z = zb*nh + zh and per-operand strides.
template<bool RELU, bool BIAS, bool RESID, bool O32, bool O16>
__global__ __launch_bounds__(32) void gemm_wmma_kernel(
    const _Float16* __restrict__ A, const _Float16* __restrict__ W,
    const float* __restrict__ bias, const float* __restrict__ resid,
    float* __restrict__ C32, _Float16* __restrict__ C16,
    int M, int N, int K, int ldc, int nh,
    long sAb, long sAh, long sWb, long sWh, long sCb, long sCh)
{
  const int z  = blockIdx.z;
  const int zb = z / nh, zh = z % nh;
  A += (long)zb * sAb + (long)zh * sAh;
  W += (long)zb * sWb + (long)zh * sWh;
  const long coff = (long)zb * sCb + (long)zh * sCh;

  const int lane = threadIdx.x & 31;
  const int l16  = lane & 15;
  const int hi   = lane >> 4;          // half-wave select
  const int m0   = blockIdx.y * 64;
  const int n0   = blockIdx.x * 64;

  v8f acc[4][4] = {};                  // 16 accumulator tiles (fp32)

  // A frag (16x32 f16): lanes 0-15 hold K 0-15, lanes 16-31 K 16-31
  const int aSel = hi * 16;
  // B frag (32x16 f16): lanes 0-15 hold K 0-15, lanes 16-31 K 16-31 (col = l16)
  const int bSel = hi * 16;

  for (int k0 = 0; k0 < K; k0 += 32) {
    v16h af[4], bf[4];
#pragma unroll
    for (int i = 0; i < 4; ++i) {
      const _Float16* ap = A + (long)(m0 + 16 * i + l16) * K + (k0 + aSel);
      v8h lo = *(const v8h*)(ap);          // 16B aligned
      v8h hv = *(const v8h*)(ap + 8);
      af[i] = __builtin_shufflevector(lo, hv, 0,1,2,3,4,5,6,7,
                                              8,9,10,11,12,13,14,15);
    }
#pragma unroll
    for (int j = 0; j < 4; ++j) {
      const _Float16* wp = W + (long)(n0 + 16 * j + l16) * K + (k0 + bSel);
      bf[j] = *(const v16h*)(wp);          // 32B aligned contiguous row chunk
    }
#pragma unroll
    for (int i = 0; i < 4; ++i)
#pragma unroll
      for (int j = 0; j < 4; ++j)
        acc[i][j] = __builtin_amdgcn_wmma_f32_16x16x32_f16(
            false, af[i], false, bf[j], (short)0, acc[i][j], false, false);
  }

  // Epilogue: C/D layout -> element (m,n): vgpr r = m%8, lane = n%16 + (m>=8)*16
#pragma unroll
  for (int j = 0; j < 4; ++j) {
    const int n = n0 + 16 * j + l16;
    const float bv = BIAS ? bias[n] : 0.0f;
#pragma unroll
    for (int i = 0; i < 4; ++i) {
#pragma unroll
      for (int r = 0; r < 8; ++r) {
        const int m = m0 + 16 * i + r + hi * 8;
        const long off = coff + (long)m * ldc + n;
        float v = acc[i][j][r] + bv;
        if (RESID) v += resid[off];
        if (RELU)  v = v > 0.0f ? v : 0.0f;
        if (O32) C32[off] = v;
        if (O16) C16[off] = (_Float16)v;
      }
    }
  }
}

// ---------------- embedding: x = wte[idx] + wpe[t], fp32 + fp16 copies -----
__global__ __launch_bounds__(256) void embed_kernel(
    const int* __restrict__ idx, const float* __restrict__ wte,
    const float* __restrict__ wpe, float* __restrict__ x32,
    _Float16* __restrict__ x16, int T, int D, long total)
{
  long i = (long)blockIdx.x * 256 + threadIdx.x;
  if (i >= total) return;
  long bt = i / D;
  int  d  = (int)(i - bt * D);
  int  t  = (int)(bt % T);
  float v = wte[(long)idx[bt] * D + d] + wpe[(long)t * D + d];
  x32[i] = v;
  x16[i] = (_Float16)v;
}

// ---------------- fp32 -> fp16 convert -------------------------------------
__global__ __launch_bounds__(256) void f32_to_f16_kernel(
    const float* __restrict__ in, _Float16* __restrict__ out, long n)
{
  long i = (long)blockIdx.x * 256 + threadIdx.x;
  if (i < n) out[i] = (_Float16)in[i];
}

// ---------------- split qkv -> q16,k16 [B,H,T,64], vT16 [B,H,64,T] ---------
__global__ __launch_bounds__(256) void split_qkv_kernel(
    const float* __restrict__ qkv, _Float16* __restrict__ q16,
    _Float16* __restrict__ k16, _Float16* __restrict__ vT16,
    int B, int T, int D, int H, long total)
{
  long i = (long)blockIdx.x * 256 + threadIdx.x;
  if (i >= total) return;
  long bt = i / D;
  int  d  = (int)(i - bt * D);
  int  b  = (int)(bt / T);
  int  t  = (int)(bt - (long)b * T);
  int  h  = d >> 6;
  int  e  = d & 63;
  const float* row = qkv + bt * (3 * D);
  long qoff = (((long)b * H + h) * T + t) * 64 + e;
  q16[qoff] = (_Float16)row[d];
  k16[qoff] = (_Float16)row[D + d];
  vT16[(((long)b * H + h) * 64 + e) * T + t] = (_Float16)row[2 * D + d];
}

// ---------------- causal softmax over keys, fp32 scores -> fp16 probs ------
__global__ __launch_bounds__(256) void softmax_causal_kernel(
    const float* __restrict__ scores, _Float16* __restrict__ probs,
    int T, float scale)
{
  const int q = blockIdx.x;
  const long base = ((long)blockIdx.y * T + q) * (long)T;
  const float* row = scores + base;
  _Float16* prow = probs + base;
  const int tid = threadIdx.x;
  __shared__ float red[256];

  float mx = -3.4e38f;
  for (int k = tid; k < T; k += 256)
    if (k <= q) mx = fmaxf(mx, row[k] * scale);
  red[tid] = mx; __syncthreads();
  for (int s = 128; s > 0; s >>= 1) {
    if (tid < s) red[tid] = fmaxf(red[tid], red[tid + s]);
    __syncthreads();
  }
  mx = red[0]; __syncthreads();

  float sum = 0.0f;
  for (int k = tid; k < T; k += 256)
    if (k <= q) sum += __expf(row[k] * scale - mx);
  red[tid] = sum; __syncthreads();
  for (int s = 128; s > 0; s >>= 1) {
    if (tid < s) red[tid] += red[tid + s];
    __syncthreads();
  }
  const float inv = 1.0f / red[0];

  for (int k = tid; k < T; k += 256) {
    float p = (k <= q) ? __expf(row[k] * scale - mx) * inv : 0.0f;
    prow[k] = (_Float16)p;
  }
}

// ---------------- LayerNorm, fused fp32 + fp16 outputs ---------------------
__global__ __launch_bounds__(256) void layernorm_kernel(
    const float* __restrict__ in, const float* __restrict__ w,
    const float* __restrict__ bb, float* __restrict__ out32,
    _Float16* __restrict__ out16, int D)
{
  const long row = blockIdx.x;
  const float* x = in + row * D;
  const int tid = threadIdx.x;
  const int per = D / 256;         // D==512 -> 2
  float vals[8];
  float s = 0.0f, ss = 0.0f;
  for (int e = 0; e < per; ++e) {
    float v = x[tid + e * 256];
    vals[e] = v; s += v; ss += v * v;
  }
  __shared__ float r1[256], r2[256];
  r1[tid] = s; r2[tid] = ss; __syncthreads();
  for (int st = 128; st > 0; st >>= 1) {
    if (tid < st) { r1[tid] += r1[tid + st]; r2[tid] += r2[tid + st]; }
    __syncthreads();
  }
  const float mean = r1[0] / (float)D;
  const float var  = r2[0] / (float)D - mean * mean;
  const float rstd = rsqrtf(var + 1e-5f);
  for (int e = 0; e < per; ++e) {
    int d = tid + e * 256;
    float o = (vals[e] - mean) * rstd * w[d] + bb[d];
    if (out32) out32[row * D + d] = o;
    if (out16) out16[row * D + d] = (_Float16)o;
  }
}

// ---------------- host-side GEMM dispatch (static epilogue configs) --------
struct GemmArgs {
  const _Float16 *A, *W;
  const float *bias, *resid;
  float *C32; _Float16 *C16;
  int M, N, K, ldc, Z, nh;
  long sAb, sAh, sWb, sWh, sCb, sCh;
};

template<bool RELU, bool BIAS, bool RESID, bool O32, bool O16>
static void gemm_launch(hipStream_t s, const GemmArgs& a)
{
  dim3 g((unsigned)(a.N / 64), (unsigned)(a.M / 64), (unsigned)a.Z), b(32, 1, 1);
  gemm_wmma_kernel<RELU, BIAS, RESID, O32, O16><<<g, b, 0, s>>>(
      a.A, a.W, a.bias, a.resid, a.C32, a.C16,
      a.M, a.N, a.K, a.ldc, a.nh, a.sAb, a.sAh, a.sWb, a.sWh, a.sCb, a.sCh);
}

extern "C" void kernel_launch(void* const* d_in, const int* in_sizes, int n_in,
                              void* d_out, int out_size, void* d_ws,
                              size_t ws_size, hipStream_t stream)
{
  (void)in_sizes; (void)n_in; (void)out_size; (void)ws_size;
  const int B = 4, T = 768, D = 512, L = 6, V = 50304, F = 2048, H = 8;
  const long BT = (long)B * T;

  const int*   idx      = (const int*)  d_in[0];
  const float* wte      = (const float*)d_in[1];
  const float* wpe      = (const float*)d_in[2];
  const float* c_attn_w = (const float*)d_in[3];
  const float* c_attn_b = (const float*)d_in[4];
  const float* c_proj_w = (const float*)d_in[5];
  const float* c_proj_b = (const float*)d_in[6];
  const float* fc_w     = (const float*)d_in[7];
  const float* fc_b     = (const float*)d_in[8];
  const float* fc2_w    = (const float*)d_in[9];
  const float* fc2_b    = (const float*)d_in[10];
  const float* ln1_w    = (const float*)d_in[11];
  const float* ln1_b    = (const float*)d_in[12];
  const float* ln2_w    = (const float*)d_in[13];
  const float* ln2_b    = (const float*)d_in[14];
  const float* lnf_w    = (const float*)d_in[15];
  const float* lnf_b    = (const float*)d_in[16];
  float* logits = (float*)d_out;

  // ---- workspace carve (256B aligned) ----
  char* p = (char*)d_ws;
  auto alloc = [&](size_t bytes) -> void* {
    void* r = (void*)p;
    p += (bytes + 255) & ~(size_t)255;
    return r;
  };
  float*    x32     = (float*)   alloc(BT * D * 4);
  _Float16* x16     = (_Float16*)alloc(BT * D * 2);
  float*    xr      = (float*)   alloc(BT * D * 4);
  float*    qkv32   = (float*)   alloc(BT * 3 * D * 4);
  _Float16* q16     = (_Float16*)alloc(BT * D * 2);
  _Float16* k16     = (_Float16*)alloc(BT * D * 2);
  _Float16* vT16    = (_Float16*)alloc(BT * D * 2);
  float*    scores  = (float*)   alloc((long)B * H * T * T * 4);
  _Float16* probs16 = (_Float16*)alloc((long)B * H * T * T * 2);
  _Float16* y16     = (_Float16*)alloc(BT * D * 2);
  _Float16* h16     = (_Float16*)alloc(BT * F * 2);
  _Float16* wte16   = (_Float16*)alloc((long)V * D * 2);
  _Float16* aw16    = (_Float16*)alloc((long)3 * D * D * 2);
  _Float16* pw16    = (_Float16*)alloc((long)D * D * 2);
  _Float16* fw16    = (_Float16*)alloc((long)F * D * 2);
  _Float16* f2w16   = (_Float16*)alloc((long)D * F * 2);

  // ---- embedding + tied-head weight conversion ----
  {
    long tot = BT * D;
    embed_kernel<<<dim3((unsigned)(tot / 256)), dim3(256), 0, stream>>>(
        idx, wte, wpe, x32, x16, T, D, tot);
    long nv = (long)V * D;
    f32_to_f16_kernel<<<dim3((unsigned)(nv / 256)), dim3(256), 0, stream>>>(
        wte, wte16, nv);
  }

  const long sQh = (long)T * 64, sQb = (long)H * T * 64;       // q16/k16/vT16
  const long sSh = (long)T * T,  sSb = (long)H * T * T;        // scores/probs

  for (int l = 0; l < L; ++l) {
    const float* aw  = c_attn_w + (long)l * 3 * D * D;
    const float* ab  = c_attn_b + (long)l * 3 * D;
    const float* pw  = c_proj_w + (long)l * D * D;
    const float* pb  = c_proj_b + (long)l * D;
    const float* fw  = fc_w     + (long)l * F * D;
    const float* fb  = fc_b     + (long)l * F;
    const float* f2w = fc2_w    + (long)l * D * F;
    const float* f2b = fc2_b    + (long)l * D;
    const float* l1w = ln1_w + (long)l * D, *l1b = ln1_b + (long)l * D;
    const float* l2w = ln2_w + (long)l * D, *l2b = ln2_b + (long)l * D;

    // weight conversion for this layer
    f32_to_f16_kernel<<<dim3((unsigned)((3L*D*D)/256)), dim3(256), 0, stream>>>(aw,  aw16,  3L*D*D);
    f32_to_f16_kernel<<<dim3((unsigned)(((long)D*D)/256)), dim3(256), 0, stream>>>(pw,  pw16,  (long)D*D);
    f32_to_f16_kernel<<<dim3((unsigned)(((long)F*D)/256)), dim3(256), 0, stream>>>(fw,  fw16,  (long)F*D);
    f32_to_f16_kernel<<<dim3((unsigned)(((long)D*F)/256)), dim3(256), 0, stream>>>(f2w, f2w16, (long)D*F);

    // qkv = x @ aw^T + ab                 [3072 x 1536]
    gemm_launch<false, true, false, true, false>(stream,
        {x16, aw16, ab, nullptr, qkv32, nullptr,
         (int)BT, 3 * D, D, 3 * D, 1, 1, 0, 0, 0, 0, 0, 0});
    split_qkv_kernel<<<dim3((unsigned)((BT * D) / 256)), dim3(256), 0, stream>>>(
        qkv32, q16, k16, vT16, B, T, D, H, BT * D);

    // scores = q @ k^T  per (b,h)         [768 x 768] x32 batches
    gemm_launch<false, false, false, true, false>(stream,
        {q16, k16, nullptr, nullptr, scores, nullptr,
         T, T, 64, T, B * H, H, sQb, sQh, sQb, sQh, sSb, sSh});
    softmax_causal_kernel<<<dim3((unsigned)T, (unsigned)(B * H)), dim3(256),
                            0, stream>>>(scores, probs16, T, 0.125f);

    // y = probs @ v  per (b,h), written into [B,T,D] as fp16
    gemm_launch<false, false, false, false, true>(stream,
        {probs16, vT16, nullptr, nullptr, nullptr, y16,
         T, 64, T, D, B * H, H, sSb, sSh, sQb, sQh, (long)T * D, 64});

    // x = LN1(x + y @ pw^T + pb)
    gemm_launch<false, true, true, true, false>(stream,
        {y16, pw16, pb, x32, xr, nullptr,
         (int)BT, D, D, D, 1, 1, 0, 0, 0, 0, 0, 0});
    layernorm_kernel<<<dim3((unsigned)BT), dim3(256), 0, stream>>>(
        xr, l1w, l1b, x32, x16, D);

    // h = relu(x @ fw^T + fb)  (fp16 only)
    gemm_launch<true, true, false, false, true>(stream,
        {x16, fw16, fb, nullptr, nullptr, h16,
         (int)BT, F, D, F, 1, 1, 0, 0, 0, 0, 0, 0});
    // x = LN2(x + h @ f2w^T + f2b)
    gemm_launch<false, true, true, true, false>(stream,
        {h16, f2w16, f2b, x32, xr, nullptr,
         (int)BT, D, F, D, 1, 1, 0, 0, 0, 0, 0, 0});
    layernorm_kernel<<<dim3((unsigned)BT), dim3(256), 0, stream>>>(
        xr, l2w, l2b, x32, x16, D);
  }

  // final LN (in-place safe: values cached in registers before store)
  layernorm_kernel<<<dim3((unsigned)BT), dim3(256), 0, stream>>>(
      x32, lnf_w, lnf_b, x32, x16, D);

  // logits = x @ wte^T                    [3072 x 50304]
  gemm_launch<false, false, false, true, false>(stream,
      {x16, wte16, nullptr, nullptr, logits, nullptr,
       (int)BT, V, D, V, 1, 1, 0, 0, 0, 0, 0, 0});
}
